// AGAOperator_34136400069082
// MI455X (gfx1250) — matrix operand order repaired and checked by reading, not compiled, not run on hardware
//
#include <hip/hip_runtime.h>
#include <hip/hip_bf16.h>
#include <math.h>

// ---------------------------------------------------------------------------
// CDNA5 WMMA (wave32): D(f32 16x16) = A(bf16 16x32) x B(bf16 32x16) + C
// ---------------------------------------------------------------------------
typedef __attribute__((ext_vector_type(16))) __bf16 v16bf;
typedef __attribute__((ext_vector_type(8)))  __bf16 v8bf;
typedef __attribute__((ext_vector_type(4)))  __bf16 v4bf;
typedef __attribute__((ext_vector_type(8)))  float  v8f;

#define TOKENS 8192   // B*S = 4*2048
#define HDIM   2048
#define BDIM   256
#define NSLOT  256
#define VBDIM  256
#define TOPK   8

static __device__ inline v8f wmma_bf16(v16bf a, v16bf b, v8f c) {
    return __builtin_amdgcn_wmma_f32_16x16x32_bf16(false, a, false, b, (short)0, c, false, false);
}

// ---- A fragment (16x32) from bf16 row-major A: two b128 loads per lane -----
// lane half = lane>>4 ; m = lane&15.
// elements 0..7  : K = k_base + half*8 + 0..7
// elements 8..15 : K = k_base + 16 + half*8 + 0..7
static __device__ inline v16bf load_a(const __bf16* A, int lda, int m_base, int k_base, int lane) {
    const int half = (lane >> 4) & 1;
    const __bf16* p = A + (size_t)(m_base + (lane & 15)) * lda + k_base + half * 8;
    v8bf lo = *(const v8bf*)(p);
    v8bf hi = *(const v8bf*)(p + 16);
    return __builtin_shufflevector(lo, hi, 0,1,2,3,4,5,6,7,8,9,10,11,12,13,14,15);
}

// ---- B fragment (32x16) from K-major bf16 weights Wt[k][n]: strided gathers -
// n = lane&15; elements 0..7: K = k_base + half*8 + 0..7, 8..15: K = k_base + 16 + half*8 + 0..7.
static __device__ inline v16bf load_b(const __bf16* Wt, int ldn, int n_base, int k_base, int lane) {
    const int half = (lane >> 4) & 1;
    const __bf16* p = Wt + (size_t)(k_base + half * 8) * ldn + n_base + (lane & 15);
    v16bf b;
#pragma unroll
    for (int s = 0; s < 8; ++s) {
        b[s]     = p[(size_t)s * ldn];
        b[8 + s] = p[(size_t)(16 + s) * ldn];
    }
    return b;
}

static __device__ inline float gelu_tanh(float x) {
    float x3 = x * x * x;
    float t = tanhf(0.7978845608028654f * (x + 0.044715f * x3));
    return 0.5f * x * (1.0f + t);
}

// ===========================================================================
// Prep A: elementwise fp32 -> bf16 (hidden), float4-vectorized.
// ===========================================================================
__global__ __launch_bounds__(256) void k_cvt_bf16(const float* __restrict__ in,
                                                  __bf16* __restrict__ out, int n4) {
    int i = blockIdx.x * blockDim.x + threadIdx.x;
    if (i < n4) {
        float4 v = ((const float4*)in)[i];
        v4bf o; o[0]=(__bf16)v.x; o[1]=(__bf16)v.y; o[2]=(__bf16)v.z; o[3]=(__bf16)v.w;
        ((v4bf*)out)[i] = o;
    }
}

// ===========================================================================
// Prep B: tiled transpose + convert: W[N][K] fp32 -> Wt[K][N] bf16.
// block (32,8); grid (K/32, N/32). Both sides coalesced via LDS.
// ===========================================================================
__global__ __launch_bounds__(256) void k_transpose_cvt(const float* __restrict__ W,
                                                       __bf16* __restrict__ Wt,
                                                       int N, int K) {
    __shared__ __bf16 t[32][33];
    const int k0 = blockIdx.x * 32, n0 = blockIdx.y * 32;
    const int tx = threadIdx.x, ty = threadIdx.y;
#pragma unroll
    for (int i = 0; i < 4; ++i) {
        int n = n0 + ty + i * 8;
        t[ty + i * 8][tx] = (__bf16)W[(size_t)n * K + k0 + tx];
    }
    __syncthreads();
#pragma unroll
    for (int i = 0; i < 4; ++i) {
        int k = k0 + ty + i * 8;
        Wt[(size_t)k * N + n0 + tx] = t[tx][ty + i * 8];
    }
}

// ===========================================================================
// Kernel 1: Q = hidden @ q_proj_w^T  [8192,2048]x[2048,256] -> bf16 [8192,256]
// 1 wave = one 16x32 output strip (2 N-tiles share the A fragment).
// grid: 512 blocks x 8 waves = 4096 waves = 512 m-tiles x 8 n-pairs.
// ===========================================================================
__global__ __launch_bounds__(256) void k_qproj(const __bf16* __restrict__ hbf,
                                               const __bf16* __restrict__ qwT,
                                               __bf16* __restrict__ qout) {
    const int lane = threadIdx.x & 31, wave = threadIdx.x >> 5;
    const int tile = blockIdx.x * 8 + wave;
    const int m_base = (tile >> 3) * 16;
    const int n_base = (tile & 7) * 32;
    v8f acc0 = {}, acc1 = {};
    for (int k = 0; k < HDIM; k += 32) {
        __builtin_prefetch(hbf + (size_t)(m_base + (lane & 15)) * HDIM + k + 32, 0, 1);
        __builtin_prefetch(qwT + (size_t)(k + 32 + lane) * BDIM + n_base, 0, 1);
        v16bf a  = load_a(hbf, HDIM, m_base, k, lane);
        v16bf b0 = load_b(qwT, BDIM, n_base,      k, lane);
        v16bf b1 = load_b(qwT, BDIM, n_base + 16, k, lane);
        acc0 = wmma_bf16(a, b0, acc0);
        acc1 = wmma_bf16(a, b1, acc1);
    }
    const int nc = lane & 15;
    const int m0 = m_base + ((lane >> 4) & 1) * 8;
#pragma unroll
    for (int r = 0; r < 8; ++r) {
        qout[(size_t)(m0 + r) * BDIM + n_base + nc]      = (__bf16)acc0[r];
        qout[(size_t)(m0 + r) * BDIM + n_base + 16 + nc] = (__bf16)acc1[r];
    }
}

// ===========================================================================
// Kernel 2: scores = Q @ slot_keys^T / 16 + log(rel); top-8 + softmax + gate.
// grid: 512 blocks (one 16-token tile) x 8 waves; wave w covers N 32w..32w+31.
// ===========================================================================
__global__ __launch_bounds__(256) void k_scores_topk(const __bf16* __restrict__ q,
                                                     const __bf16* __restrict__ skT,
                                                     const float* __restrict__ reliability,
                                                     int* __restrict__ top_idx,
                                                     float* __restrict__ attn_w,
                                                     float* __restrict__ gate_out) {
    __shared__ float sc[16][NSLOT + 1];
    const int lane = threadIdx.x & 31, wave = threadIdx.x >> 5;
    const int m_base = blockIdx.x * 16;
    const int n_base = wave * 32;
    v8f acc0 = {}, acc1 = {};
    for (int k = 0; k < BDIM; k += 32) {
        v16bf a  = load_a(q, BDIM, m_base, k, lane);
        v16bf b0 = load_b(skT, NSLOT, n_base,      k, lane);
        v16bf b1 = load_b(skT, NSLOT, n_base + 16, k, lane);
        acc0 = wmma_bf16(a, b0, acc0);
        acc1 = wmma_bf16(a, b1, acc1);
    }
    const int nc = lane & 15;
    const int half = (lane >> 4) & 1;
    const float rel0 = logf(fmaxf(reliability[n_base + nc],      1e-10f));
    const float rel1 = logf(fmaxf(reliability[n_base + 16 + nc], 1e-10f));
#pragma unroll
    for (int r = 0; r < 8; ++r) {
        sc[half * 8 + r][n_base + nc]      = acc0[r] * 0.0625f + rel0;
        sc[half * 8 + r][n_base + 16 + nc] = acc1[r] * 0.0625f + rel1;
    }
    __syncthreads();
    if (threadIdx.x < 16) {
        const int m = threadIdx.x;
        float bs[TOPK]; int bi[TOPK];
#pragma unroll
        for (int j = 0; j < TOPK; ++j) { bs[j] = -3.402823466e38f; bi[j] = 0; }
        for (int n = 0; n < NSLOT; ++n) {
            float s = sc[m][n];
            if (s > bs[TOPK - 1]) {               // strict '>' keeps earliest index on ties
                int j = TOPK - 1;
                while (j > 0 && s > bs[j - 1]) { bs[j] = bs[j - 1]; bi[j] = bi[j - 1]; --j; }
                bs[j] = s; bi[j] = n;
            }
        }
        const float mx = bs[0];
        float w[TOPK], sum = 0.0f;
#pragma unroll
        for (int j = 0; j < TOPK; ++j) { w[j] = expf(bs[j] - mx); sum += w[j]; }
        const float inv = 1.0f / sum;
        const int tok = m_base + m;
#pragma unroll
        for (int j = 0; j < TOPK; ++j) {
            attn_w[(size_t)tok * TOPK + j] = w[j] * inv;
            top_idx[(size_t)tok * TOPK + j] = bi[j];
        }
        gate_out[tok] = 1.0f / (1.0f + expf(-bs[0]));
    }
}

// ===========================================================================
// Kernel 3: aux[tok] = sum_k attn_k * slot_values[idx_k]  -> bf16 [8192,2048]
// grid: 8192 blocks x 256 thr. slot_values (2MB) is L2 resident.
// ===========================================================================
__global__ __launch_bounds__(256) void k_gather(const float* __restrict__ slot_values,
                                                const int* __restrict__ top_idx,
                                                const float* __restrict__ attn_w,
                                                __bf16* __restrict__ aux) {
    const int tok = blockIdx.x;
    __shared__ float w[TOPK];
    __shared__ int   id[TOPK];
    if (threadIdx.x < TOPK) {
        w[threadIdx.x]  = attn_w[(size_t)tok * TOPK + threadIdx.x];
        id[threadIdx.x] = top_idx[(size_t)tok * TOPK + threadIdx.x];
    }
    __syncthreads();
    for (int h = threadIdx.x * 4; h < HDIM; h += 256 * 4) {
        float4 s = {0.f, 0.f, 0.f, 0.f};
#pragma unroll
        for (int k = 0; k < TOPK; ++k) {
            const float4 v = *(const float4*)(slot_values + (size_t)id[k] * HDIM + h);
            s.x += w[k] * v.x; s.y += w[k] * v.y; s.z += w[k] * v.z; s.w += w[k] * v.w;
        }
        v4bf o; o[0]=(__bf16)s.x; o[1]=(__bf16)s.y; o[2]=(__bf16)s.z; o[3]=(__bf16)s.w;
        *(v4bf*)(aux + (size_t)tok * HDIM + h) = o;
    }
}

// ===========================================================================
// Kernel 4: mid = gelu(aux @ value_down_w^T) [8192,2048]x[2048,256] -> bf16
// Same tiling as k_qproj.
// ===========================================================================
__global__ __launch_bounds__(256) void k_down_gelu(const __bf16* __restrict__ aux,
                                                   const __bf16* __restrict__ dwT,
                                                   __bf16* __restrict__ mid) {
    const int lane = threadIdx.x & 31, wave = threadIdx.x >> 5;
    const int tile = blockIdx.x * 8 + wave;
    const int m_base = (tile >> 3) * 16;
    const int n_base = (tile & 7) * 32;
    v8f acc0 = {}, acc1 = {};
    for (int k = 0; k < HDIM; k += 32) {
        __builtin_prefetch(aux + (size_t)(m_base + (lane & 15)) * HDIM + k + 32, 0, 1);
        __builtin_prefetch(dwT + (size_t)(k + 32 + lane) * VBDIM + n_base, 0, 1);
        v16bf a  = load_a(aux, HDIM, m_base, k, lane);
        v16bf b0 = load_b(dwT, VBDIM, n_base,      k, lane);
        v16bf b1 = load_b(dwT, VBDIM, n_base + 16, k, lane);
        acc0 = wmma_bf16(a, b0, acc0);
        acc1 = wmma_bf16(a, b1, acc1);
    }
    const int nc = lane & 15;
    const int m0 = m_base + ((lane >> 4) & 1) * 8;
#pragma unroll
    for (int r = 0; r < 8; ++r) {
        mid[(size_t)(m0 + r) * VBDIM + n_base + nc]      = (__bf16)gelu_tanh(acc0[r]);
        mid[(size_t)(m0 + r) * VBDIM + n_base + 16 + nc] = (__bf16)gelu_tanh(acc1[r]);
    }
}

// ===========================================================================
// Kernel 5: out = primary + gate * (mid @ value_up_w^T)
//           [8192,256]x[256,2048] -> f32 [8192,2048]
// 1 wave = one 16x64 strip (4 N-tiles share A). 512 mt x 32 nquad = 16384
// waves = 2048 blocks.
// ===========================================================================
__global__ __launch_bounds__(256) void k_up_gate(const __bf16* __restrict__ mid,
                                                 const __bf16* __restrict__ uwT,
                                                 const float* __restrict__ primary,
                                                 const float* __restrict__ gate_in,
                                                 float* __restrict__ out) {
    const int lane = threadIdx.x & 31, wave = threadIdx.x >> 5;
    const int tile = blockIdx.x * 8 + wave;
    const int m_base = (tile >> 5) * 16;
    const int n_base = (tile & 31) * 64;
    v8f acc0 = {}, acc1 = {}, acc2 = {}, acc3 = {};
    for (int k = 0; k < VBDIM; k += 32) {
        v16bf a  = load_a(mid, VBDIM, m_base, k, lane);
        v16bf b0 = load_b(uwT, HDIM, n_base,      k, lane);
        v16bf b1 = load_b(uwT, HDIM, n_base + 16, k, lane);
        v16bf b2 = load_b(uwT, HDIM, n_base + 32, k, lane);
        v16bf b3 = load_b(uwT, HDIM, n_base + 48, k, lane);
        acc0 = wmma_bf16(a, b0, acc0);
        acc1 = wmma_bf16(a, b1, acc1);
        acc2 = wmma_bf16(a, b2, acc2);
        acc3 = wmma_bf16(a, b3, acc3);
    }
    const int nc = lane & 15;
    const int m0 = m_base + ((lane >> 4) & 1) * 8;
#pragma unroll
    for (int r = 0; r < 8; ++r) {
        const int m = m0 + r;
        const float g = gate_in[m];
        const size_t row = (size_t)m * HDIM + n_base + nc;
        out[row]      = primary[row]      + g * acc0[r];
        out[row + 16] = primary[row + 16] + g * acc1[r];
        out[row + 32] = primary[row + 32] + g * acc2[r];
        out[row + 48] = primary[row + 48] + g * acc3[r];
    }
}

// ===========================================================================
extern "C" void kernel_launch(void* const* d_in, const int* in_sizes, int n_in,
                              void* d_out, int out_size, void* d_ws, size_t ws_size,
                              hipStream_t stream) {
    const float* hidden   = (const float*)d_in[0];   // [4,2048,2048]
    const float* primary  = (const float*)d_in[1];   // [4,2048,2048]
    const float* qw       = (const float*)d_in[2];   // [256,2048]
    const float* skeys    = (const float*)d_in[3];   // [256,256]
    const float* svals    = (const float*)d_in[4];   // [256,2048]
    const float* rel      = (const float*)d_in[5];   // [256]
    const float* dw       = (const float*)d_in[6];   // [256,2048]
    const float* uw       = (const float*)d_in[7];   // [2048,256]
    float* out = (float*)d_out;

    char* p = (char*)d_ws;
    __bf16* hbf  = (__bf16*)p; p += (size_t)TOKENS * HDIM  * sizeof(__bf16);  // 32 MB
    __bf16* aux  = (__bf16*)p; p += (size_t)TOKENS * HDIM  * sizeof(__bf16);  // 32 MB
    __bf16* qbf  = (__bf16*)p; p += (size_t)TOKENS * BDIM  * sizeof(__bf16);  //  4 MB
    __bf16* mid  = (__bf16*)p; p += (size_t)TOKENS * VBDIM * sizeof(__bf16);  //  4 MB
    __bf16* qwT  = (__bf16*)p; p += (size_t)HDIM * BDIM  * sizeof(__bf16);    //  1 MB
    __bf16* dwT  = (__bf16*)p; p += (size_t)HDIM * VBDIM * sizeof(__bf16);    //  1 MB
    __bf16* uwT  = (__bf16*)p; p += (size_t)VBDIM * HDIM * sizeof(__bf16);    //  1 MB
    __bf16* skT  = (__bf16*)p; p += (size_t)BDIM * NSLOT * sizeof(__bf16);    // 128 KB
    int*    tidx = (int*)p;    p += (size_t)TOKENS * TOPK * sizeof(int);
    float*  attn = (float*)p;  p += (size_t)TOKENS * TOPK * sizeof(float);
    float*  gate = (float*)p;  p += (size_t)TOKENS * sizeof(float);

    // --- prep: bf16 activation copy + K-major bf16 weights ---
    k_cvt_bf16<<<(TOKENS * HDIM / 4 + 255) / 256, 256, 0, stream>>>(hidden, hbf, TOKENS * HDIM / 4);
    {
        dim3 blk(32, 8);
        k_transpose_cvt<<<dim3(HDIM / 32,  BDIM / 32),  blk, 0, stream>>>(qw,    qwT, BDIM,  HDIM);
        k_transpose_cvt<<<dim3(BDIM / 32,  NSLOT / 32), blk, 0, stream>>>(skeys, skT, NSLOT, BDIM);
        k_transpose_cvt<<<dim3(HDIM / 32,  VBDIM / 32), blk, 0, stream>>>(dw,    dwT, VBDIM, HDIM);
        k_transpose_cvt<<<dim3(VBDIM / 32, HDIM / 32),  blk, 0, stream>>>(uw,    uwT, HDIM,  VBDIM);
    }
    // --- main pipeline ---
    k_qproj      <<< 512, 256, 0, stream>>>(hbf, qwT, qbf);
    k_scores_topk<<< 512, 256, 0, stream>>>(qbf, skT, rel, tidx, attn, gate);
    k_gather     <<<8192, 256, 0, stream>>>(svals, tidx, attn, aux);
    k_down_gelu  <<< 512, 256, 0, stream>>>(aux, dwT, mid);
    k_up_gate    <<<2048, 256, 0, stream>>>(mid, uwT, primary, gate, out);
}
